// DummyLinear4bit_8057358647560
// MI455X (gfx1250) — matrix operand order repaired and checked by reading, not compiled, not run on hardware
//
#include <hip/hip_runtime.h>
#include <stdint.h>

// ---------------------------------------------------------------------------
// NF4-dequant fused GEMM for MI455X (gfx1250, wave32, WMMA + async-to-LDS)
//   out[t, o] = sum_k x[t,k] * W[o,k],  W dequantized on the fly from nibbles
// Pipeline (1 barrier / k-tile):
//   sA (x tile)  : double-buffered, filled via global_load_async_to_lds_b128
//   sB (W tile)  : double-buffered, register-staged packed bytes -> dequant
//                  VALU co-executes with the WMMA block, stores ds_store_b128
// ---------------------------------------------------------------------------

#define TOKENS 8192
#define IN_F   4096
#define OUT_F  4096

typedef _Float16 half_t;
typedef half_t v2h  __attribute__((ext_vector_type(2)));
typedef __fp16 v2hf __attribute__((ext_vector_type(2)));
typedef half_t v16h __attribute__((ext_vector_type(16)));
typedef float  v8f  __attribute__((ext_vector_type(8)));

#define M_TILE 128
#define N_TILE 128
#define K_TILE 64
#define LDS_STRIDE 72                 // 64 halves + 8 pad (144 B, 16-B aligned)
#define NT (IN_F / K_TILE)            // 64 k-tiles

union FragU { v16h v; uint4 q[2]; };
union H8   { half_t h[8]; v2h h2[4]; uint4 q; };

struct WStage {                       // register-staged packed W + scales
    int4  b[4];                       // 16 "bytes" (stored one per int32)
    float s, o;                       // w = q * s - o
};

__device__ __forceinline__ float fast_rcp(float v) {
#if __has_builtin(__builtin_amdgcn_rcpf)
    return __builtin_amdgcn_rcpf(v);        // v_rcp_f32
#else
    return 1.0f / v;
#endif
}

__device__ __forceinline__ v2h pack_f16(float lo, float hi) {
#if __has_builtin(__builtin_amdgcn_cvt_pkrtz)
    v2hf t = __builtin_amdgcn_cvt_pkrtz(lo, hi);   // v_cvt_pk_rtz_f16_f32
    return __builtin_bit_cast(v2h, t);
#else
    v2h r; r.x = (half_t)lo; r.y = (half_t)hi; return r;
#endif
}

__device__ __forceinline__ uint32_t lds_off(const half_t* p) {
    // generic pointers to LDS carry the LDS offset in addr[31:0]
    return (uint32_t)(uintptr_t)p;
}

__device__ __forceinline__ void async_copy_b128(uint32_t dst_lds, const half_t* src) {
    asm volatile("global_load_async_to_lds_b128 %0, %1, off"
                 :: "v"(dst_lds), "v"((unsigned long long)(uintptr_t)src)
                 : "memory");
}

__device__ __forceinline__ void wait_async0() {
#if __has_builtin(__builtin_amdgcn_s_wait_asynccnt)
    __builtin_amdgcn_s_wait_asynccnt(0);
#else
    asm volatile("s_wait_asynccnt 0x0" ::: "memory");
#endif
}

__device__ __forceinline__ void loadA_async(const half_t* __restrict__ x,
                                            int tBase, int k0,
                                            half_t* sAbuf, int tid) {
    #pragma unroll
    for (int i = 0; i < 4; ++i) {
        int v   = tid + i * 256;           // vec8 slot (0..1023)
        int row = v >> 3;                  // 0..127
        int cv  = (v & 7) * 8;             // half offset, multiple of 8
        async_copy_b128(lds_off(&sAbuf[row * LDS_STRIDE + cv]),
                        x + (size_t)(tBase + row) * IN_F + k0 + cv);
    }
}

__device__ __forceinline__ void loadW(WStage& w,
                                      const int* __restrict__ qw,
                                      const int* __restrict__ absmax1,
                                      const float* __restrict__ code1,
                                      const float* __restrict__ off1,
                                      const float* __restrict__ absmax2,
                                      const float* __restrict__ code2,
                                      int nBase, int k0, int tid) {
    int row   = tid >> 1;                 // out-feature within tile
    int halfp = tid & 1;                  // which 32-elem half of the row
    size_t L  = (size_t)(nBase + row) * IN_F + k0;   // linear elem idx, %64==0
    int b1 = (int)(L >> 6);               // 64-elem primary block
    int b2 = (int)(L >> 8);               // 256-elem secondary block
    const int4* qp = reinterpret_cast<const int4*>(qw + (L >> 1) + halfp * 16);
    #pragma unroll
    for (int j = 0; j < 4; ++j) w.b[j] = qp[j];
    float s1 = (float)absmax1[b1] * fast_rcp(code1[b1]);
    float s2 = absmax2[b2] * fast_rcp(code2[b2]);
    w.s = s1 * s2;
    w.o = off1[b1] * w.s;
}

__device__ __forceinline__ void storeW(const WStage& w, half_t* sBbuf, int tid) {
    int row = tid >> 1, halfp = tid & 1;
    half_t* dst = &sBbuf[row * LDS_STRIDE + halfp * 32];
    #pragma unroll
    for (int j = 0; j < 4; ++j) {
        int vals[4] = {w.b[j].x, w.b[j].y, w.b[j].z, w.b[j].w};
        H8 h8;
        #pragma unroll
        for (int e = 0; e < 4; ++e) {
            float lo = __builtin_fmaf((float)(vals[e] & 15),        w.s, -w.o);
            float hi = __builtin_fmaf((float)((vals[e] >> 4) & 15), w.s, -w.o);
            h8.h2[e] = pack_f16(lo, hi);          // one v_cvt_pk_rtz_f16_f32
        }
        *reinterpret_cast<uint4*>(dst + j * 8) = h8.q;   // one ds_store_b128
    }
}

__global__ __launch_bounds__(256)
void nf4_wmma_gemm(const half_t* __restrict__ x,
                   const int*    __restrict__ qw,
                   const int*    __restrict__ absmax1,
                   const float*  __restrict__ code1,
                   const float*  __restrict__ off1,
                   const float*  __restrict__ absmax2,
                   const float*  __restrict__ code2,
                   float*        __restrict__ out)
{
    __shared__ half_t sA[2][M_TILE * LDS_STRIDE];   // x tile  x2  (36 KB)
    __shared__ half_t sB[2][N_TILE * LDS_STRIDE];   // W tile  x2  (36 KB)

    const int tid  = threadIdx.x;
    const int lane = tid & 31;
    const int wave = tid >> 5;
    const int waveM = (wave >> 1) * 32;   // 0,32,64,96
    const int waveN = (wave & 1)  * 64;   // 0,64
    const int lHi = lane >> 4;            // half-wave
    const int lLo = lane & 15;

    const int nBase = blockIdx.x * N_TILE;
    const int tBase = blockIdx.y * M_TILE;

    v8f acc[2][4];
    #pragma unroll
    for (int mi = 0; mi < 2; ++mi)
        #pragma unroll
        for (int ni = 0; ni < 4; ++ni)
            acc[mi][ni] = (v8f){0.f, 0.f, 0.f, 0.f, 0.f, 0.f, 0.f, 0.f};

    WStage w;

    // ---- prologue: stage tile 0 ------------------------------------------
    loadA_async(x, tBase, 0, sA[0], tid);
    loadW(w, qw, absmax1, code1, off1, absmax2, code2, nBase, 0, tid);
    storeW(w, sB[0], tid);
    wait_async0();
    __syncthreads();

    for (int t = 0; t < NT; ++t) {
        const int buf = t & 1;

        // ---- issue next tile's traffic before compute --------------------
        if (t + 1 < NT) {
            loadA_async(x, tBase, (t + 1) * K_TILE, sA[buf ^ 1], tid);
            loadW(w, qw, absmax1, code1, off1, absmax2, code2,
                  nBase, (t + 1) * K_TILE, tid);
        }

        // ---- WMMA over K_TILE (two k-steps of 32) ------------------------
        #pragma unroll
        for (int kk = 0; kk < K_TILE; kk += 32) {
            FragU afrag[2];
            #pragma unroll
            for (int mi = 0; mi < 2; ++mi) {
                const half_t* base =
                    &sA[buf][(waveM + mi * 16 + lLo) * LDS_STRIDE + kk + lHi * 8];
                afrag[mi].q[0] = *reinterpret_cast<const uint4*>(base);        // K +0..7
                afrag[mi].q[1] = *reinterpret_cast<const uint4*>(base + 16);   // K +16..23
            }
            FragU bfrag[4];
            #pragma unroll
            for (int ni = 0; ni < 4; ++ni) {
                const half_t* base =
                    &sB[buf][(waveN + ni * 16 + lLo) * LDS_STRIDE + kk + lHi * 8];
                bfrag[ni].q[0] = *reinterpret_cast<const uint4*>(base);
                bfrag[ni].q[1] = *reinterpret_cast<const uint4*>(base + 16);
            }
            #pragma unroll
            for (int mi = 0; mi < 2; ++mi)
                #pragma unroll
                for (int ni = 0; ni < 4; ++ni)
                    acc[mi][ni] = __builtin_amdgcn_wmma_f32_16x16x32_f16(
                        false, afrag[mi].v, false, bfrag[ni].v,
                        (short)0, acc[mi][ni], false, false);
        }

        // ---- dequant W(t+1) into the other buffer (VALU overlaps XDL) ----
        if (t + 1 < NT)
            storeW(w, sB[buf ^ 1], tid);

        wait_async0();                    // sA[buf^1] fully written by this wave
        __syncthreads();                  // publish sA[buf^1] + sB[buf^1]
    }

    // ---- epilogue: C layout VGPR r <-> M = r + 8*lHi, N = lLo -------------
    #pragma unroll
    for (int mi = 0; mi < 2; ++mi) {
        #pragma unroll
        for (int ni = 0; ni < 4; ++ni) {
            #pragma unroll
            for (int r = 0; r < 8; ++r) {
                int m = tBase + waveM + mi * 16 + r + lHi * 8;
                int n = nBase + waveN + ni * 16 + lLo;
                out[(size_t)m * OUT_F + n] = acc[mi][ni][r];
            }
        }
    }
}

extern "C" void kernel_launch(void* const* d_in, const int* in_sizes, int n_in,
                              void* d_out, int out_size, void* d_ws, size_t ws_size,
                              hipStream_t stream) {
    const half_t* x       = (const half_t*)d_in[0];
    const int*    qweight = (const int*)  d_in[1];
    const int*    absmax1 = (const int*)  d_in[2];
    const float*  code1   = (const float*)d_in[3];
    const float*  off1    = (const float*)d_in[4];
    const float*  absmax2 = (const float*)d_in[5];
    const float*  code2   = (const float*)d_in[6];
    float*        out     = (float*)d_out;

    dim3 grid(OUT_F / N_TILE, TOKENS / M_TILE);   // 32 x 64
    dim3 block(256);
    nf4_wmma_gemm<<<grid, block, 0, stream>>>(x, qweight, absmax1, code1, off1,
                                              absmax2, code2, out);
}
